// Deeplab_ResNet_Backbone_32375463477657
// MI455X (gfx1250) — matrix operand order repaired and checked by reading, not compiled, not run on hardware
//
#include <hip/hip_runtime.h>
#include <hip/hip_bf16.h>
#include <string.h>

// ---------------------------------------------------------------------------
// GCN-ResNet backbone for MI455X (gfx1250, wave32, WMMA).
//   * GEMMs: v_wmma_f32_16x16x32_bf16, f32 accumulate. K padded to x32 so the
//     inner loop is guard-free; double-buffered LDS A-tile -> ONE barrier per
//     K-step; B pre-packed to bf16 in wmma B-operand order (32B/lane loads).
//   * Aggregation: L2-resident f32 atomic scatter, 16 features per edge-load.
// ---------------------------------------------------------------------------

#define NN 50000     // nodes  (50000 = 16 * 3125, tile-exact)
#define EE 400000    // edges before self-loops

typedef __attribute__((ext_vector_type(4)))  __bf16 v4bf;
typedef __attribute__((ext_vector_type(8)))  __bf16 v8bf;
typedef __attribute__((ext_vector_type(16))) __bf16 v16bf;
typedef __attribute__((ext_vector_type(8)))  float  v8f;

// f32 -> bf16, round-to-nearest-even
__device__ __forceinline__ __bf16 f2bf(float f) {
    union { float f; unsigned u; } x; x.f = f;
    unsigned r = x.u + 0x7fffu + ((x.u >> 16) & 1u);
    unsigned short h = (unsigned short)(r >> 16);
    __bf16 out;
    __builtin_memcpy(&out, &h, 2);
    return out;
}

__device__ __forceinline__ v4bf f2bf4(float4 v) {
    v4bf t; t.x = f2bf(v.x); t.y = f2bf(v.y); t.z = f2bf(v.z); t.w = f2bf(v.w);
    return t;
}

// ------------------------------ small utility kernels ----------------------

__global__ void fill_f32(float* __restrict__ p, long long n, float v) {
    long long i = (long long)blockIdx.x * blockDim.x + threadIdx.x;
    if (i < n) p[i] = v;
}

__global__ void deg_edges(const int* __restrict__ dst, float* __restrict__ deg, int e) {
    int i = blockIdx.x * blockDim.x + threadIdx.x;
    if (i < e) atomicAdd(&deg[dst[i]], 1.0f);
}

__global__ void rsqrt_k(const float* __restrict__ deg, float* __restrict__ dinv, int n) {
    int i = blockIdx.x * blockDim.x + threadIdx.x;
    if (i < n) {
        float d = deg[i];
        dinv[i] = (d > 0.0f) ? rsqrtf(d) : 0.0f;
    }
}

__global__ void norm_k(const int* __restrict__ src, const int* __restrict__ dst,
                       const float* __restrict__ dinv, float* __restrict__ norm,
                       int e, int n) {
    int i = blockIdx.x * blockDim.x + threadIdx.x;
    if (i < e) {
        norm[i] = dinv[src[i]] * dinv[dst[i]];
    } else if (i < e + n) {
        float d = dinv[i - e];
        norm[i] = d * d;               // self-loop
    }
}

// zero-pad x [NN,39] -> xp [NN,64] so every GEMM K is a multiple of 32
__global__ void pad_x64(const float* __restrict__ x, float* __restrict__ xp) {
    int i = blockIdx.x * blockDim.x + threadIdx.x;
    if (i >= NN * 64) return;
    int r = i >> 6, c = i & 63;
    xp[i] = (c < 39) ? x[(size_t)r * 39 + c] : 0.0f;
}

// Pack W [Kreal, C] f32 -> Wp bf16 in wmma B-order: [Kpad/32][C][32]
__global__ void pack_w_bf16(const float* __restrict__ W, __bf16* __restrict__ Wp,
                            int Kreal, int Kpad, int C) {
    int idx = blockIdx.x * blockDim.x + threadIdx.x;
    int total = (Kpad >> 5) * C;
    if (idx >= total) return;
    int blk = idx / C;
    int n   = idx - blk * C;
    __bf16* out = Wp + ((size_t)blk * C + n) * 32;
    int k0 = blk << 5;
#pragma unroll
    for (int i = 0; i < 32; ++i) {
        int k = k0 + i;
        out[i] = f2bf((k < Kreal) ? W[(size_t)k * C + n] : 0.0f);
    }
}

// ------------------------------ WMMA GEMM ----------------------------------
// out[NN, C] = H[NN, K] @ W[K, C],  K % 32 == 0, C % 64 == 0.
// Block = 128 threads (4 waves); block tile 16 rows x 64 cols, one 16x16
// wmma tile per wave. Double-buffered LDS A-tile: one barrier per K-step,
// next global float4 issued before the wmma (software pipelined).
#define LDS_ROW 40   // bf16 elements per LDS row (80 B stride, 16B-aligned)
__global__ void __launch_bounds__(128)
gcn_gemm_wmma(const float* __restrict__ H, const __bf16* __restrict__ Wp,
              float* __restrict__ out, int K, int C) {
    __shared__ __bf16 Atile[2][16 * LDS_ROW];

    const int tid  = threadIdx.x;
    const int wave = tid >> 5;
    const int lane = tid & 31;
    const int rowBase = blockIdx.x * 16;
    const int colBase = blockIdx.y * 64 + wave * 16;

    const int m16 = lane & 15;
    const int hi  = lane >> 4;              // 0: low K half, 1: high K half
    const int n   = colBase + m16;

    // A staging: each thread owns one float4 of the 16x32 tile
    const int ar = tid >> 3;
    const int ac = (tid & 7) << 2;
    const float* hsrc = H + (size_t)(rowBase + ar) * K + ac;
    v4bf* ldst0 = reinterpret_cast<v4bf*>(&Atile[0][ar * LDS_ROW + ac]);
    v4bf* ldst1 = reinterpret_cast<v4bf*>(&Atile[1][ar * LDS_ROW + ac]);

    // A fetch: row m16, v8bf chunks 2*hi and 2*hi+1 (K {0..15}/{16..31})
    const v8bf* arow0 = reinterpret_cast<const v8bf*>(&Atile[0][m16 * LDS_ROW]);
    const v8bf* arow1 = reinterpret_cast<const v8bf*>(&Atile[1][m16 * LDS_ROW]);

    // B fetch: column n, contiguous 16 bf16 at K-offset hi*16 within block
    const __bf16* wsrc = Wp + (size_t)n * 32 + hi * 16;
    const size_t wstep = (size_t)C * 32;    // per K-block advance

    v8f acc = {};
    const int nkb = K >> 5;                 // >= 2 always

    // prologue: stage tile 0
    *ldst0 = f2bf4(*reinterpret_cast<const float4*>(hsrc));

    for (int kb = 0; kb < nkb - 1; ++kb) {
        __syncthreads();
        // issue next A global load early
        float4 hn = *reinterpret_cast<const float4*>(hsrc + (size_t)(kb + 1) * 32);
        const v8bf* arow = (kb & 1) ? arow1 : arow0;
        v8bf a0 = arow[2 * hi];
        v8bf a1 = arow[2 * hi + 1];
        v16bf a = __builtin_shufflevector(a0, a1, 0, 1, 2, 3, 4, 5, 6, 7,
                                                  8, 9, 10, 11, 12, 13, 14, 15);
        v16bf b = *reinterpret_cast<const v16bf*>(wsrc + (size_t)kb * wstep);
        acc = __builtin_amdgcn_wmma_f32_16x16x32_bf16(
            false, a, false, b, (short)0, acc, false, false);
        // stage tile kb+1 into the other buffer (reads of it were pre-barrier)
        *((kb & 1) ? ldst0 : ldst1) = f2bf4(hn);
    }

    // epilogue: last K-step
    {
        __syncthreads();
        const int kb = nkb - 1;
        const v8bf* arow = (kb & 1) ? arow1 : arow0;
        v8bf a0 = arow[2 * hi];
        v8bf a1 = arow[2 * hi + 1];
        v16bf a = __builtin_shufflevector(a0, a1, 0, 1, 2, 3, 4, 5, 6, 7,
                                                  8, 9, 10, 11, 12, 13, 14, 15);
        v16bf b = *reinterpret_cast<const v16bf*>(wsrc + (size_t)kb * wstep);
        acc = __builtin_amdgcn_wmma_f32_16x16x32_bf16(
            false, a, false, b, (short)0, acc, false, false);
    }

    // C/D layout: VGPR v -> row = v + hi*8, col = lane&15
#pragma unroll
    for (int v = 0; v < 8; ++v) {
        out[(size_t)(rowBase + v + hi * 8) * C + n] = acc[v];
    }
}

// ------------------------------ edge scatter -------------------------------
// agg[dst] += msg[src] * norm over E real edges + N self-loops.
// 16 features per thread: one src/dst/norm load feeds 4 float4 gathers +
// 16 L2-resident f32 atomics (4x less index traffic than 4-wide).
__global__ void gcn_scatter(const float* __restrict__ msg,
                            const int* __restrict__ src, const int* __restrict__ dst,
                            const float* __restrict__ norm,
                            float* __restrict__ agg, int etot, int eedges, int C) {
    int e = blockIdx.x * blockDim.x + threadIdx.x;
    if (e >= etot) return;
    if (e + 16384 < eedges) {               // stream edge lists ahead of use
        __builtin_prefetch(src + e + 16384, 0, 0);
        __builtin_prefetch(dst + e + 16384, 0, 0);
    }
    int s, d;
    if (e < eedges) { s = src[e]; d = dst[e]; }
    else            { s = e - eedges; d = s; }
    float nm = norm[e];
    int cc = blockIdx.y << 4;
    const float4* vin = reinterpret_cast<const float4*>(msg + (size_t)s * C + cc);
    float* o = agg + (size_t)d * C + cc;
#pragma unroll
    for (int q = 0; q < 4; ++q) {
        float4 v = vin[q];
        atomicAdd(o + 4 * q + 0, v.x * nm);
        atomicAdd(o + 4 * q + 1, v.y * nm);
        atomicAdd(o + 4 * q + 2, v.z * nm);
        atomicAdd(o + 4 * q + 3, v.w * nm);
    }
}

// ------------------------------ fused epilogue -----------------------------
// out = relu(agg + b [+ b2] [+ addend])   (C is a power of two)
__global__ void finalize_k(const float* __restrict__ agg, const float* __restrict__ b,
                           const float* __restrict__ b2, const float* __restrict__ add,
                           float* __restrict__ out, long long total, int C) {
    long long i = (long long)blockIdx.x * blockDim.x + threadIdx.x;
    if (i >= total) return;
    int c = (int)(i & (long long)(C - 1));
    float v = agg[i] + b[c];
    if (b2)  v += b2[c];
    if (add) v += add[i];
    out[i] = fmaxf(v, 0.0f);
}

// ---------------------------------------------------------------------------

extern "C" void kernel_launch(void* const* d_in, const int* in_sizes, int n_in,
                              void* d_out, int out_size, void* d_ws, size_t ws_size,
                              hipStream_t stream) {
    (void)in_sizes; (void)n_in; (void)out_size; (void)ws_size;

    const float* x    = (const float*)d_in[0];
    const int*   ei   = (const int*)d_in[1];
    const int*   esrc = ei;
    const int*   edst = ei + EE;

    const float* P[24];
    for (int i = 0; i < 24; ++i) P[i] = (const float*)d_in[2 + i];
    // P: 0 W_seed 1 b_seed 2 W00 3 b00 4 W01 5 b01
    //    6 Wd1 7 bd1 8 W10 9 b10 10 W11 11 b11
    //   12 Wd2 13 bd2 14 W20 15 b20 16 W21 17 b21
    //   18 Wd3 19 bd3 20 W30 21 b30 22 W31 23 b31

    // workspace layout (floats); big buffers 32B aligned
    float*  ws    = (float*)d_ws;
    float*  deg   = ws;                     // NN
    float*  dinv  = deg + NN;               // NN
    float*  norm  = dinv + NN;              // EE + NN
    float*  xpad  = norm + (EE + NN);       // NN * 64
    __bf16* wp    = (__bf16*)(xpad + (size_t)NN * 64);   // 512*512 bf16
    float*  tmp   = (float*)(wp + 512 * 512);            // NN*512 messages
    const size_t big = (size_t)NN * 512;
    float*  agg   = tmp + big;              // NN*512
    float*  hB    = agg + big;              // NN*512
    float*  resid = hB + big;               // NN*512
    float*  hA    = (float*)d_out;          // ping-pong partner; final here

    const int ETOT = EE + NN;

    // ---- normalization precompute (deg includes self-loop) ----
    fill_f32<<<(NN + 255) / 256, 256, 0, stream>>>(deg, NN, 1.0f);
    deg_edges<<<(EE + 255) / 256, 256, 0, stream>>>(edst, deg, EE);
    rsqrt_k<<<(NN + 255) / 256, 256, 0, stream>>>(deg, dinv, NN);
    norm_k<<<(ETOT + 255) / 256, 256, 0, stream>>>(esrc, edst, dinv, norm, EE, NN);

    // ---- pad x once: every GEMM K becomes a multiple of 32 ----
    pad_x64<<<(NN * 64 + 255) / 256, 256, 0, stream>>>(x, xpad);

    // one GCN aggregation: aggdst = scatter(norm * (hin @ W))
    auto agg_from = [&](const float* hin, const float* W, int Kreal, int Kpad,
                        int C, float* aggdst) {
        int pk = (Kpad >> 5) * C;
        pack_w_bf16<<<(pk + 255) / 256, 256, 0, stream>>>(W, wp, Kreal, Kpad, C);
        dim3 gg(NN / 16, (unsigned)(C / 64));
        gcn_gemm_wmma<<<gg, 128, 0, stream>>>(hin, wp, tmp, Kpad, C);
        long long tot = (long long)NN * C;
        fill_f32<<<(unsigned)((tot + 255) / 256), 256, 0, stream>>>(aggdst, tot, 0.0f);
        dim3 gs((ETOT + 255) / 256, (unsigned)(C / 16));
        gcn_scatter<<<gs, 256, 0, stream>>>(tmp, esrc, edst, norm, aggdst, ETOT, EE, C);
    };
    auto fin = [&](const float* aggp, const float* b, const float* b2,
                   const float* add, float* outp, int C) {
        long long tot = (long long)NN * C;
        finalize_k<<<(unsigned)((tot + 255) / 256), 256, 0, stream>>>(
            aggp, b, b2, add, outp, tot, C);
    };

    // ---- seed: h = relu(gcn(x, W_seed) + b_seed) ----
    agg_from(xpad, P[0], 39, 64, 64, agg);
    fin(agg, P[1], nullptr, nullptr, hA, 64);

    // ---- segment 0 (64->64, identity residual) ----
    agg_from(hA, P[2], 64, 64, 64, agg);  fin(agg, P[3], nullptr, hA, hB, 64);
    agg_from(hB, P[4], 64, 64, 64, agg);  fin(agg, P[5], nullptr, hB, hA, 64);

    // ---- segment 1 (64->128, GCN downsample residual) ----
    agg_from(hA, P[6],  64, 64, 128, resid);                           // Wd1
    agg_from(hA, P[8],  64, 64, 128, agg);   fin(agg, P[9],  P[7], resid, hB, 128);
    agg_from(hB, P[10], 128, 128, 128, agg); fin(agg, P[11], nullptr, hB, hA, 128);

    // ---- segment 2 (128->256) ----
    agg_from(hA, P[12], 128, 128, 256, resid);                         // Wd2
    agg_from(hA, P[14], 128, 128, 256, agg); fin(agg, P[15], P[13], resid, hB, 256);
    agg_from(hB, P[16], 256, 256, 256, agg); fin(agg, P[17], nullptr, hB, hA, 256);

    // ---- segment 3 (256->512) ----
    agg_from(hA, P[18], 256, 256, 512, resid);                         // Wd3
    agg_from(hA, P[20], 256, 256, 512, agg); fin(agg, P[21], P[19], resid, hB, 512);
    agg_from(hB, P[22], 512, 512, 512, agg); fin(agg, P[23], nullptr, hB, hA, 512);
    // hA == d_out : final [50000, 512] activation lands in the output buffer.
}
